// SSDMixer_43628277793675
// MI455X (gfx1250) — compile-verified, matrix operand order checked by reading, not run on hardware
//
#include <hip/hip_runtime.h>
#include <hip/hip_bf16.h>

// ---------------- constants (from reference) ----------------
#define BSZ      2
#define LSEQ     4096
#define MTOK     (BSZ * LSEQ)        // 8192
#define DIMK     2048
#define D_INNER  4096
#define NHEADS   64
#define HEADDIM  64
#define D_STATE  64
#define NGROUPS  1
#define CHUNKQ   64
#define NCHUNK   (LSEQ / CHUNKQ)     // 64
#define DPROJ    (2 * D_INNER + 2 * NGROUPS * D_STATE + NHEADS) // 8384
#define RMS_EPS  1.1920929e-07f
#define EXP_CLAMP 10000.0f

typedef __attribute__((ext_vector_type(16))) __bf16           v16bf;
typedef __attribute__((ext_vector_type(8)))  float            v8f;
typedef __attribute__((ext_vector_type(8)))  unsigned short   v8us;
typedef __attribute__((ext_vector_type(16))) unsigned short   v16us;
typedef __attribute__((ext_vector_type(4)))  unsigned int     v4u;
typedef __attribute__((ext_vector_type(8)))  int              v8i;
typedef __attribute__((ext_vector_type(4)))  int              v4i;

__device__ __forceinline__ unsigned short f2bf(float f) {
    unsigned u = __builtin_bit_cast(unsigned, f);
    unsigned r = u + 0x7FFFu + ((u >> 16) & 1u);   // round-to-nearest-even
    return (unsigned short)(r >> 16);
}
__device__ __forceinline__ float bf2f(unsigned short u) {
    unsigned v = ((unsigned)u) << 16;
    return __builtin_bit_cast(float, v);
}

__device__ __forceinline__ v8f wmma_bf16(v16bf a, v16bf b, v8f c) {
    return __builtin_amdgcn_wmma_f32_16x16x32_bf16(false, a, false, b, (short)0, c, false, false);
}

// Load a 16x32 bf16 fragment (A-layout; B operand uses same layout from an
// N-major [N][K] matrix). p points at tile origin, ld = row stride in elems.
__device__ __forceinline__ v16bf frag_ld(const unsigned short* p, int ld) {
    int lane = threadIdx.x & 31;
    const unsigned short* r = p + (size_t)(lane & 15) * ld + ((lane >> 4) << 3);
    v8us lo = *(const v8us*)r;
    v8us hi = *(const v8us*)(r + 16);
    v16us t;
#pragma unroll
    for (int i = 0; i < 8; ++i) { t[i] = lo[i]; t[i + 8] = hi[i]; }
    return __builtin_bit_cast(v16bf, t);
}

// ---------------- TDM: 1-D contiguous bf16 tile, global -> LDS ----------------
// D# per CDNA5 ISA §8.3-8.6: group0 {count=1, lds_addr, global_addr, type=2},
// group1 {data_size=2B, tensor_dim0=tile_dim0=nelems, tensor_dim1=tile_dim1=1}.
__device__ __forceinline__ void tdm_load_lds_1d(unsigned lds_off, const void* gsrc, int nelems) {
    unsigned long long ga = (unsigned long long)(size_t)gsrc;
    v4u g0;
    g0[0] = 1u;                                        // count=1, gather=0
    g0[1] = lds_off;                                   // LDS byte address
    g0[2] = (unsigned)ga;                              // global_addr[31:0]
    g0[3] = (unsigned)((ga >> 32) & 0x01ffffffull) | (2u << 30);  // addr[56:32] | type=2
    v8i g1;
    g1[0] = (int)(1u << 16);                           // data_size=1 (2 bytes)
    g1[1] = (int)(((unsigned)nelems & 0xffffu) << 16); // tensor_dim0[15:0] @bits63:48
    g1[2] = (int)((((unsigned)nelems >> 16) & 0xffffu) | (1u << 16)); // dim0 hi | tensor_dim1=1
    g1[3] = (int)(((unsigned)nelems & 0xffffu) << 16); // tile_dim0 @bits127:112
    g1[4] = 1;                                         // tile_dim1=1, tile_dim2=0
    g1[5] = nelems;                                    // tensor_dim0_stride lo32
    g1[6] = 0;
    g1[7] = 0;
    v4i gz = {0, 0, 0, 0};
#if defined(__clang_major__) && (__clang_major__ >= 23)
    v8i gz8 = {0, 0, 0, 0, 0, 0, 0, 0};
    __builtin_amdgcn_tensor_load_to_lds(g0, g1, gz, gz, gz8, 0);
#else
    __builtin_amdgcn_tensor_load_to_lds(g0, g1, gz, gz, 0);
#endif
}
__device__ __forceinline__ unsigned lds_addr_of(const void* p) {
    return (unsigned)(size_t)p;   // LDS generic addr: low 32 bits = LDS offset
}

// ---------------- K0: fq4 quantize (group 128) -> bf16 ----------------
__global__ void __launch_bounds__(128) k_quant_fq4(const float* __restrict__ w,
                                                   unsigned short* __restrict__ out) {
    __shared__ float red[4];
    int g = blockIdx.x, t = threadIdx.x;
    float v = w[(size_t)g * 128 + t];
    float a = fabsf(v);
#pragma unroll
    for (int o = 16; o > 0; o >>= 1) a = fmaxf(a, __shfl_xor(a, o, 32));
    if ((t & 31) == 0) red[t >> 5] = a;
    __syncthreads();
    float m = fmaxf(fmaxf(red[0], red[1]), fmaxf(red[2], red[3]));
    float scale = fmaxf(m, 1e-10f) / 7.0f;
    float q = fminf(fmaxf(rintf(v / scale), -8.f), 7.f) * scale;
    out[(size_t)g * 128 + t] = f2bf(q);
}

// ---------------- K1: f32 -> bf16 ----------------
__global__ void __launch_bounds__(256) k_cvt_bf16(const float* __restrict__ in,
                                                  unsigned short* __restrict__ out, size_t n) {
    size_t i = (size_t)blockIdx.x * 256 + threadIdx.x;
    if (i < n) out[i] = f2bf(in[i]);
}

// ---------------- K2a: GEMM 32x32/wave (small N) ----------------
__global__ void __launch_bounds__(128) k_gemm_wmma(const unsigned short* __restrict__ A,
                                                   const unsigned short* __restrict__ W,
                                                   float* __restrict__ C,
                                                   int M, int N, int K) {
    int wave   = blockIdx.x * 4 + (threadIdx.x >> 5);
    int tilesN = N >> 5;
    int wm = (wave / tilesN) << 5;
    int wn = (wave % tilesN) << 5;
    if (wm >= M) return;
    v8f acc[2][2] = {};
    const unsigned short* a0p = A + (size_t)wm * K;
    const unsigned short* w0p = W + (size_t)wn * K;
    for (int k = 0; k < K; k += 32) {
        __builtin_prefetch((const void*)(a0p + k + 256), 0, 1);
        __builtin_prefetch((const void*)(w0p + k + 256), 0, 1);
        v16bf a0 = frag_ld(a0p + k, K);
        v16bf a1 = frag_ld(a0p + (size_t)16 * K + k, K);
        v16bf b0 = frag_ld(w0p + k, K);
        v16bf b1 = frag_ld(w0p + (size_t)16 * K + k, K);
        acc[0][0] = wmma_bf16(a0, b0, acc[0][0]);
        acc[0][1] = wmma_bf16(a0, b1, acc[0][1]);
        acc[1][0] = wmma_bf16(a1, b0, acc[1][0]);
        acc[1][1] = wmma_bf16(a1, b1, acc[1][1]);
    }
    int lane = threadIdx.x & 31;
    int ro = (lane >> 4) << 3, co = lane & 15;
#pragma unroll
    for (int ti = 0; ti < 2; ++ti)
#pragma unroll
        for (int tj = 0; tj < 2; ++tj) {
            size_t rbase = (size_t)(wm + ti * 16 + ro);
            int cc = wn + tj * 16 + co;
#pragma unroll
            for (int r = 0; r < 8; ++r)
                C[(rbase + r) * N + cc] = acc[ti][tj][r];
        }
}

// ---------------- K2b: GEMM 64x64/wave (big GEMMs, 32 FLOP/B reuse) ----------------
__global__ void __launch_bounds__(128) k_gemm_wmma64(const unsigned short* __restrict__ A,
                                                     const unsigned short* __restrict__ W,
                                                     float* __restrict__ C,
                                                     int M, int N, int K) {
    int wave   = blockIdx.x * 4 + (threadIdx.x >> 5);
    int tilesN = N >> 6;
    int wm = (wave / tilesN) << 6;
    int wn = (wave % tilesN) << 6;
    if (wm >= M) return;
    v8f acc[4][4] = {};
    const unsigned short* ap = A + (size_t)wm * K;
    const unsigned short* wp = W + (size_t)wn * K;
    for (int k = 0; k < K; k += 32) {
        __builtin_prefetch((const void*)(ap + k + 256), 0, 1);
        __builtin_prefetch((const void*)(wp + k + 256), 0, 1);
        v16bf a[4], b[4];
#pragma unroll
        for (int i = 0; i < 4; ++i) {
            a[i] = frag_ld(ap + (size_t)(16 * i) * K + k, K);
            b[i] = frag_ld(wp + (size_t)(16 * i) * K + k, K);
        }
#pragma unroll
        for (int i = 0; i < 4; ++i)
#pragma unroll
            for (int j = 0; j < 4; ++j)
                acc[i][j] = wmma_bf16(a[i], b[j], acc[i][j]);
    }
    int lane = threadIdx.x & 31;
    int ro = (lane >> 4) << 3, co = lane & 15;
#pragma unroll
    for (int ti = 0; ti < 4; ++ti)
#pragma unroll
        for (int tj = 0; tj < 4; ++tj) {
            size_t rbase = (size_t)(wm + ti * 16 + ro);
            int cc = wn + tj * 16 + co;
#pragma unroll
            for (int r = 0; r < 8; ++r)
                C[(rbase + r) * N + cc] = acc[ti][tj][r];
        }
}

// ---------------- K3a: per-token head params ----------------
__global__ void __launch_bounds__(64) k_head_params(const float* __restrict__ zx,
                                                    const float* __restrict__ lam_o,
                                                    const float* __restrict__ A_log,
                                                    const float* __restrict__ dt_bias,
                                                    float* __restrict__ A_disc,
                                                    float* __restrict__ gammaB,
                                                    float* __restrict__ betaB,
                                                    float* __restrict__ dt_avg) {
    __shared__ float red[2];
    int tok = blockIdx.x, h = threadIdx.x;
    float dtr = zx[(size_t)tok * DPROJ + 2 * D_INNER + 2 * D_STATE + h] + dt_bias[h];
    float dt  = dtr > 20.f ? dtr : log1pf(expf(dtr));
    float ad  = -expf(A_log[h]) * dt;
    float lv  = 1.f / (1.f + expf(-lam_o[(size_t)tok * NHEADS + h]));
    A_disc[(size_t)tok * NHEADS + h] = ad;
    gammaB[(size_t)tok * NHEADS + h] = lv * dt;
    betaB [(size_t)tok * NHEADS + h] = (1.f - lv) * dt * expf(ad);
    float s = dt;
#pragma unroll
    for (int o = 16; o > 0; o >>= 1) s += __shfl_xor(s, o, 32);
    if ((h & 31) == 0) red[h >> 5] = s;
    __syncthreads();
    if (h == 0) dt_avg[tok] = (red[0] + red[1]) / (float)NHEADS;
}

// ---------------- K3b: RoPE angle cumsum along L ----------------
__global__ void __launch_bounds__(256) k_rope_angles(const float* __restrict__ theta_o,
                                                     const float* __restrict__ dt_avg,
                                                     float* __restrict__ cosb,
                                                     float* __restrict__ sinb) {
    __shared__ float buf[256];
    int b = blockIdx.x >> 5, d = blockIdx.x & 31;
    float carry = 0.f;
    for (int base = 0; base < LSEQ; base += 256) {
        size_t tok = (size_t)b * LSEQ + base + threadIdx.x;
        float v = theta_o[tok * 32 + d] * dt_avg[tok];
        buf[threadIdx.x] = v;
        __syncthreads();
        for (int o = 1; o < 256; o <<= 1) {
            float add = (threadIdx.x >= (unsigned)o) ? buf[threadIdx.x - o] : 0.f;
            __syncthreads();
            buf[threadIdx.x] += add;
            __syncthreads();
        }
        float ang = carry + buf[threadIdx.x];
        cosb[tok * 32 + d] = cosf(ang);
        sinb[tok * 32 + d] = sinf(ang);
        carry += buf[255];
        __syncthreads();
    }
}

// ---------------- K3c: B/C rmsnorm + bias + RoPE -> bf16 ----------------
__global__ void __launch_bounds__(64) k_bc_norm_rope(const float* __restrict__ zx,
                                                     const float* __restrict__ B_bias,
                                                     const float* __restrict__ C_bias,
                                                     const float* __restrict__ cosb,
                                                     const float* __restrict__ sinb,
                                                     unsigned short* __restrict__ Bm_bf,
                                                     unsigned short* __restrict__ Cm_bf) {
    __shared__ float nb[64];
    __shared__ float red[2];
    int tok = blockIdx.x, t = threadIdx.x;
    for (int part = 0; part < 2; ++part) {
        const float* src  = zx + (size_t)tok * DPROJ + 2 * D_INNER + part * D_STATE;
        const float* bias = part ? C_bias : B_bias;
        float raw = src[t];
        float ss = raw * raw;
#pragma unroll
        for (int o = 16; o > 0; o >>= 1) ss += __shfl_xor(ss, o, 32);
        if ((t & 31) == 0) red[t >> 5] = ss;
        __syncthreads();
        float inv = rsqrtf((red[0] + red[1]) / (float)D_STATE + RMS_EPS);
        nb[t] = raw * inv + bias[t];
        __syncthreads();
        int d = t & 31;
        float c = cosb[(size_t)tok * 32 + d], s = sinb[(size_t)tok * 32 + d];
        float outv = (t < 32) ? nb[t] * c - nb[t + 32] * s
                              : nb[t - 32] * s + nb[t] * c;
        unsigned short* dstb = part ? Cm_bf : Bm_bf;
        dstb[(size_t)tok * 64 + t] = f2bf(outv);
        __syncthreads();
    }
}

// ---------------- K3d: x_heads = silu(x_conv); X (chunk-transposed bf16) ----------------
// X_bfT layout: [b][c][h][p][l]  (8 KB contiguous per (b,c,h) -> TDM-friendly)
__global__ void __launch_bounds__(256) k_build_X(const float* __restrict__ zx,
                                                 const float* __restrict__ gammaB,
                                                 const float* __restrict__ betaB,
                                                 float* __restrict__ xh,
                                                 unsigned short* __restrict__ X_bfT) {
    size_t idx = (size_t)blockIdx.x * 256 + threadIdx.x;   // over MTOK*4096
    size_t tok = idx >> 12;
    int e = (int)(idx & 4095), h = e >> 6, p = e & 63;
    int b = (int)(tok >> 12);
    int lpos = (int)(tok & (LSEQ - 1));
    int c = lpos >> 6, l = lpos & 63;
    float xv = zx[tok * DPROJ + D_INNER + e];
    float sv = xv / (1.f + expf(-xv));
    xh[idx] = sv;
    float Xv = sv * gammaB[tok * NHEADS + h];
    if (lpos > 0) {
        float pv = zx[(tok - 1) * DPROJ + D_INNER + e];
        Xv += (pv / (1.f + expf(-pv))) * betaB[tok * NHEADS + h];
    }
    X_bfT[((((size_t)b * NCHUNK + c) * NHEADS + h) * 64 + p) * 64 + l] = f2bf(Xv);
}

// ---------------- K4: intra-chunk SSD (per b,chunk,head), TDM-staged tiles ----------------
__global__ void __launch_bounds__(128) k_ssd_intra(const unsigned short* __restrict__ Bm_bf,
                                                   const unsigned short* __restrict__ Cm_bf,
                                                   const unsigned short* __restrict__ X_bfT,
                                                   const float* __restrict__ A_disc,
                                                   float* __restrict__ Y,
                                                   float* __restrict__ states,
                                                   float* __restrict__ a_chs) {
    __shared__ float Acs[64];
    __shared__ unsigned short XT[64][64];    // X^T  [p][l]   (TDM)
    __shared__ unsigned short Bt[64][64];    // B    [l][n]   (TDM)
    __shared__ unsigned short Ct[64][64];    // C    [l][n]   (TDM)
    __shared__ unsigned short BdT[64][64];   // (decay*B)^T [n][l]
    __shared__ unsigned short GL[64][64];    // (C B^T) ∘ Lmat  [l][s]
    int blk = blockIdx.x;
    int h = blk & 63, c = (blk >> 6) & (NCHUNK - 1), b = blk / (64 * NCHUNK);
    size_t tok0 = (size_t)b * LSEQ + (size_t)c * CHUNKQ;
    int wv = threadIdx.x >> 5;
    if (wv == 0) {
        tdm_load_lds_1d(lds_addr_of(&XT[0][0]),
                        X_bfT + (((size_t)b * NCHUNK + c) * NHEADS + h) * 4096, 4096);
        tdm_load_lds_1d(lds_addr_of(&Bt[0][0]), Bm_bf + tok0 * 64, 4096);
        tdm_load_lds_1d(lds_addr_of(&Ct[0][0]), Cm_bf + tok0 * 64, 4096);
    }
    if (threadIdx.x == 0) {
        float s = 0.f;
        for (int l = 0; l < 64; ++l) { s += A_disc[(tok0 + l) * NHEADS + h]; Acs[l] = s; }
        a_chs[((size_t)b * NCHUNK + c) * NHEADS + h] = s;
    }
    if (wv == 0) __builtin_amdgcn_s_wait_tensorcnt(0);
    __syncthreads();
    float Alast = Acs[63];
    for (int i = threadIdx.x; i < 4096; i += 128) {
        int l = i >> 6, n = i & 63;
        float dec = expf(Alast - Acs[l]);
        BdT[n][l] = f2bf(dec * bf2f(Bt[l][n]));
    }
    __syncthreads();
    int qm = (wv >> 1) << 5, qn = (wv & 1) << 5;
    int lane = threadIdx.x & 31;
    int ro = (lane >> 4) << 3, co = lane & 15;
    // ---- G = C @ B^T, masked by segsum decay ----
    {
        v8f acc[2][2] = {};
        for (int k = 0; k < 64; k += 32) {
            v16bf a0 = frag_ld(&Ct[qm][0] + k, 64);
            v16bf a1 = frag_ld(&Ct[qm + 16][0] + k, 64);
            v16bf b0 = frag_ld(&Bt[qn][0] + k, 64);
            v16bf b1 = frag_ld(&Bt[qn + 16][0] + k, 64);
            acc[0][0] = wmma_bf16(a0, b0, acc[0][0]);
            acc[0][1] = wmma_bf16(a0, b1, acc[0][1]);
            acc[1][0] = wmma_bf16(a1, b0, acc[1][0]);
            acc[1][1] = wmma_bf16(a1, b1, acc[1][1]);
        }
#pragma unroll
        for (int ti = 0; ti < 2; ++ti)
#pragma unroll
            for (int tj = 0; tj < 2; ++tj)
#pragma unroll
                for (int r = 0; r < 8; ++r) {
                    int l = qm + ti * 16 + ro + r, s = qn + tj * 16 + co;
                    float f = (s <= l) ? fminf(expf(Acs[l] - Acs[s]), EXP_CLAMP) : 0.f;
                    GL[l][s] = f2bf(acc[ti][tj][r] * f);
                }
    }
    __syncthreads();
    // ---- Y_diag = GL @ X   (W-operand = XT) ----
    {
        v8f acc[2][2] = {};
        for (int k = 0; k < 64; k += 32) {
            v16bf a0 = frag_ld(&GL[qm][0] + k, 64);
            v16bf a1 = frag_ld(&GL[qm + 16][0] + k, 64);
            v16bf b0 = frag_ld(&XT[qn][0] + k, 64);
            v16bf b1 = frag_ld(&XT[qn + 16][0] + k, 64);
            acc[0][0] = wmma_bf16(a0, b0, acc[0][0]);
            acc[0][1] = wmma_bf16(a0, b1, acc[0][1]);
            acc[1][0] = wmma_bf16(a1, b0, acc[1][0]);
            acc[1][1] = wmma_bf16(a1, b1, acc[1][1]);
        }
#pragma unroll
        for (int ti = 0; ti < 2; ++ti)
#pragma unroll
            for (int tj = 0; tj < 2; ++tj)
#pragma unroll
                for (int r = 0; r < 8; ++r) {
                    int l = qm + ti * 16 + ro + r, p = qn + tj * 16 + co;
                    Y[((tok0 + l) * NHEADS + h) * HEADDIM + p] = acc[ti][tj][r];
                }
    }
    // ---- states[p,n] = XT @ BdT^T ----
    {
        v8f acc[2][2] = {};
        for (int k = 0; k < 64; k += 32) {
            v16bf a0 = frag_ld(&XT[qm][0] + k, 64);
            v16bf a1 = frag_ld(&XT[qm + 16][0] + k, 64);
            v16bf b0 = frag_ld(&BdT[qn][0] + k, 64);
            v16bf b1 = frag_ld(&BdT[qn + 16][0] + k, 64);
            acc[0][0] = wmma_bf16(a0, b0, acc[0][0]);
            acc[0][1] = wmma_bf16(a0, b1, acc[0][1]);
            acc[1][0] = wmma_bf16(a1, b0, acc[1][0]);
            acc[1][1] = wmma_bf16(a1, b1, acc[1][1]);
        }
        size_t sbase = (((size_t)b * NCHUNK + c) * NHEADS + h) * 4096;
#pragma unroll
        for (int ti = 0; ti < 2; ++ti)
#pragma unroll
            for (int tj = 0; tj < 2; ++tj)
#pragma unroll
                for (int r = 0; r < 8; ++r) {
                    int p = qm + ti * 16 + ro + r, n = qn + tj * 16 + co;
                    states[sbase + (size_t)p * 64 + n] = acc[ti][tj][r];
                }
    }
}

// ---------------- K5: inter-chunk state recurrence (in place) ----------------
__global__ void __launch_bounds__(256) k_chunk_scan(float* __restrict__ states,
                                                    const float* __restrict__ a_chs) {
    int b = blockIdx.x >> 6, h = blockIdx.x & 63;
    float S[16];
#pragma unroll
    for (int i = 0; i < 16; ++i) S[i] = 0.f;
    for (int c = 0; c < NCHUNK; ++c) {
        size_t base = (((size_t)b * NCHUNK + c) * NHEADS + h) * 4096;
        float dec = fminf(expf(a_chs[((size_t)b * NCHUNK + c) * NHEADS + h]), EXP_CLAMP);
#pragma unroll
        for (int i = 0; i < 16; ++i) {
            size_t off = base + (size_t)i * 256 + threadIdx.x;
            float st = states[off];
            states[off] = S[i];            // inter[c] = new_states[c]
            S[i] = dec * S[i] + st;
        }
    }
}

// ---------------- K6: Y += exp(A_cs[l]) * (C @ inter^T) ----------------
__global__ void __launch_bounds__(128) k_ssd_inter(const unsigned short* __restrict__ Cm_bf,
                                                   const float* __restrict__ inter,
                                                   const float* __restrict__ A_disc,
                                                   float* __restrict__ Y) {
    __shared__ float Acs[64];
    __shared__ unsigned short Ct[64][64];   // C [l][n]  (TDM)
    __shared__ unsigned short IT[64][64];   // inter [p][n] bf16
    int blk = blockIdx.x;
    int h = blk & 63, c = (blk >> 6) & (NCHUNK - 1), b = blk / (64 * NCHUNK);
    size_t tok0 = (size_t)b * LSEQ + (size_t)c * CHUNKQ;
    int wv = threadIdx.x >> 5;
    if (wv == 0)
        tdm_load_lds_1d(lds_addr_of(&Ct[0][0]), Cm_bf + tok0 * 64, 4096);
    if (threadIdx.x == 0) {
        float s = 0.f;
        for (int l = 0; l < 64; ++l) { s += A_disc[(tok0 + l) * NHEADS + h]; Acs[l] = s; }
    }
    size_t sbase = (((size_t)b * NCHUNK + c) * NHEADS + h) * 4096;
    for (int i = threadIdx.x; i < 4096; i += 128)
        IT[i >> 6][i & 63] = f2bf(inter[sbase + i]);
    if (wv == 0) __builtin_amdgcn_s_wait_tensorcnt(0);
    __syncthreads();
    int qm = (wv >> 1) << 5, qn = (wv & 1) << 5;
    int lane = threadIdx.x & 31;
    int ro = (lane >> 4) << 3, co = lane & 15;
    v8f acc[2][2] = {};
    for (int k = 0; k < 64; k += 32) {
        v16bf a0 = frag_ld(&Ct[qm][0] + k, 64);
        v16bf a1 = frag_ld(&Ct[qm + 16][0] + k, 64);
        v16bf b0 = frag_ld(&IT[qn][0] + k, 64);
        v16bf b1 = frag_ld(&IT[qn + 16][0] + k, 64);
        acc[0][0] = wmma_bf16(a0, b0, acc[0][0]);
        acc[0][1] = wmma_bf16(a0, b1, acc[0][1]);
        acc[1][0] = wmma_bf16(a1, b0, acc[1][0]);
        acc[1][1] = wmma_bf16(a1, b1, acc[1][1]);
    }
#pragma unroll
    for (int ti = 0; ti < 2; ++ti)
#pragma unroll
        for (int tj = 0; tj < 2; ++tj)
#pragma unroll
            for (int r = 0; r < 8; ++r) {
                int l = qm + ti * 16 + ro + r, p = qn + tj * 16 + co;
                size_t yi = ((tok0 + l) * NHEADS + h) * HEADDIM + p;
                Y[yi] += expf(Acs[l]) * acc[ti][tj][r];
            }
}

// ---------------- K7a: gate with silu(z), +D skip, rmsnorm -> bf16 ----------------
__global__ void __launch_bounds__(256) k_finalize_norm(const float* __restrict__ Y,
                                                       const float* __restrict__ xh,
                                                       const float* __restrict__ zx,
                                                       const float* __restrict__ D_param,
                                                       unsigned short* __restrict__ ynorm) {
    __shared__ float red[8];
    int tok = blockIdx.x, t = threadIdx.x;
    float v[16];
    float ss = 0.f;
#pragma unroll
    for (int i = 0; i < 16; ++i) {
        int e = i * 256 + t, h = e >> 6;
        float z  = zx[(size_t)tok * DPROJ + e];
        float sz = z / (1.f + expf(-z));
        float y  = (Y[(size_t)tok * D_INNER + e] + xh[(size_t)tok * D_INNER + e] * D_param[h]) * sz;
        v[i] = y;
        ss += y * y;
    }
#pragma unroll
    for (int o = 16; o > 0; o >>= 1) ss += __shfl_xor(ss, o, 32);
    if ((t & 31) == 0) red[t >> 5] = ss;
    __syncthreads();
    float tot = 0.f;
#pragma unroll
    for (int i = 0; i < 8; ++i) tot += red[i];
    float inv = rsqrtf(tot / (float)D_INNER + RMS_EPS);
#pragma unroll
    for (int i = 0; i < 16; ++i)
        ynorm[(size_t)tok * D_INNER + i * 256 + t] = f2bf(v[i] * inv);
}

// ---------------- host side ----------------
extern "C" void kernel_launch(void* const* d_in, const int* in_sizes, int n_in,
                              void* d_out, int out_size, void* d_ws, size_t ws_size,
                              hipStream_t stream) {
    (void)in_sizes; (void)n_in; (void)out_size; (void)ws_size;
    const float* x_in    = (const float*)d_in[0];
    const float* W_in    = (const float*)d_in[1];
    const float* W_theta = (const float*)d_in[2];
    const float* W_lam   = (const float*)d_in[3];
    const float* W_out   = (const float*)d_in[4];
    const float* B_bias  = (const float*)d_in[5];
    const float* C_bias  = (const float*)d_in[6];
    const float* A_log   = (const float*)d_in[7];
    const float* D_param = (const float*)d_in[8];
    const float* dt_bias = (const float*)d_in[9];
    float* out = (float*)d_out;

    char* w = (char*)d_ws;
    size_t off = 0;
    auto alloc = [&](size_t bytes) -> void* {
        void* p = w + off;
        off = (off + bytes + 255) & ~(size_t)255;
        return p;
    };
    unsigned short* wq_in  = (unsigned short*)alloc((size_t)DPROJ * DIMK * 2);
    unsigned short* wq_th  = (unsigned short*)alloc((size_t)32 * DIMK * 2);
    unsigned short* wq_la  = (unsigned short*)alloc((size_t)NHEADS * DIMK * 2);
    unsigned short* wq_out = (unsigned short*)alloc((size_t)DIMK * D_INNER * 2);
    unsigned short* x_bf   = (unsigned short*)alloc((size_t)MTOK * DIMK * 2);
    float* zx      = (float*)alloc((size_t)MTOK * DPROJ * 4);
    float* theta_o = (float*)alloc((size_t)MTOK * 32 * 4);
    float* lam_o   = (float*)alloc((size_t)MTOK * NHEADS * 4);
    float* A_disc  = (float*)alloc((size_t)MTOK * NHEADS * 4);
    float* gammaB  = (float*)alloc((size_t)MTOK * NHEADS * 4);
    float* betaB   = (float*)alloc((size_t)MTOK * NHEADS * 4);
    float* dt_avg  = (float*)alloc((size_t)MTOK * 4);
    float* cosb    = (float*)alloc((size_t)MTOK * 32 * 4);
    float* sinb    = (float*)alloc((size_t)MTOK * 32 * 4);
    unsigned short* Bm_bf = (unsigned short*)alloc((size_t)MTOK * 64 * 2);
    unsigned short* Cm_bf = (unsigned short*)alloc((size_t)MTOK * 64 * 2);
    float* xh      = (float*)alloc((size_t)MTOK * D_INNER * 4);
    unsigned short* X_bfT = (unsigned short*)alloc((size_t)MTOK * D_INNER * 2);
    float* Ybuf    = (float*)alloc((size_t)MTOK * D_INNER * 4);
    float* states  = (float*)alloc((size_t)BSZ * NCHUNK * NHEADS * 4096 * 4);
    float* a_chs   = (float*)alloc((size_t)BSZ * NCHUNK * NHEADS * 4);
    unsigned short* ynorm = (unsigned short*)alloc((size_t)MTOK * D_INNER * 2);

    // K0: quantize weights
    k_quant_fq4<<<(DPROJ * DIMK) / 128, 128, 0, stream>>>(W_in, wq_in);
    k_quant_fq4<<<(32 * DIMK) / 128, 128, 0, stream>>>(W_theta, wq_th);
    k_quant_fq4<<<(NHEADS * DIMK) / 128, 128, 0, stream>>>(W_lam, wq_la);
    k_quant_fq4<<<(DIMK * D_INNER) / 128, 128, 0, stream>>>(W_out, wq_out);
    // K1: x -> bf16
    k_cvt_bf16<<<((size_t)MTOK * DIMK + 255) / 256, 256, 0, stream>>>(x_in, x_bf, (size_t)MTOK * DIMK);
    // K2: projections (64x64 wave tiles for big GEMMs)
    k_gemm_wmma64<<<(MTOK / 64) * (DPROJ / 64) / 4, 128, 0, stream>>>(x_bf, wq_in, zx, MTOK, DPROJ, DIMK);
    k_gemm_wmma  <<<(MTOK / 32) * (32 / 32) / 4, 128, 0, stream>>>(x_bf, wq_th, theta_o, MTOK, 32, DIMK);
    k_gemm_wmma64<<<(MTOK / 64) * (NHEADS / 64) / 4, 128, 0, stream>>>(x_bf, wq_la, lam_o, MTOK, NHEADS, DIMK);
    // K3: per-token nonlinearities / RoPE / X assembly
    k_head_params<<<MTOK, 64, 0, stream>>>(zx, lam_o, A_log, dt_bias, A_disc, gammaB, betaB, dt_avg);
    k_rope_angles<<<BSZ * 32, 256, 0, stream>>>(theta_o, dt_avg, cosb, sinb);
    k_bc_norm_rope<<<MTOK, 64, 0, stream>>>(zx, B_bias, C_bias, cosb, sinb, Bm_bf, Cm_bf);
    k_build_X<<<((size_t)MTOK * D_INNER) / 256, 256, 0, stream>>>(zx, gammaB, betaB, xh, X_bfT);
    // K4-K6: SSD
    k_ssd_intra<<<BSZ * NCHUNK * NHEADS, 128, 0, stream>>>(Bm_bf, Cm_bf, X_bfT, A_disc, Ybuf, states, a_chs);
    k_chunk_scan<<<BSZ * NHEADS, 256, 0, stream>>>(states, a_chs);
    k_ssd_inter<<<BSZ * NCHUNK * NHEADS, 128, 0, stream>>>(Cm_bf, states, A_disc, Ybuf);
    // K7: gate + rmsnorm + out projection
    k_finalize_norm<<<MTOK, 256, 0, stream>>>(Ybuf, xh, zx, D_param, ynorm);
    k_gemm_wmma64<<<(MTOK / 64) * (DIMK / 64) / 4, 128, 0, stream>>>(ynorm, wq_out, out, MTOK, DIMK, D_INNER);
}
